// SymbolicCausalSelfAttentionALiBi_66314295050783
// MI455X (gfx1250) — matrix operand order repaired and checked by reading, not compiled, not run on hardware
//
#include <hip/hip_runtime.h>
#include <hip/hip_bf16.h>
#include <math.h>

// Problem constants (from reference)
#define BB 2
#define TT 768
#define CC 64
#define HH 64
#define NROWS (BB * TT)   // 1536 rows of x
#define NCOLS (2 * CC)    // 128 output cols of qk (q | k)

typedef __attribute__((ext_vector_type(16))) _Float16 v16h;
typedef __attribute__((ext_vector_type(8)))  float    v8f;

// -----------------------------------------------------------------------------
// Kernel A: qk = x @ W_attn^T  via V_WMMA_F32_16X16X32_F16.
// One wave (32 threads) per 16x16 output tile; K=64 -> two WMMA ops.
// grid = (NROWS/16, NCOLS/16), block = 32 (one full wave32 -> EXEC all ones).
// -----------------------------------------------------------------------------
__global__ __launch_bounds__(32)
void qk_proj_wmma_kernel(const float* __restrict__ x,     // (1536, 64)
                         const float* __restrict__ W,     // (128, 64)
                         float* __restrict__ qk)          // (1536, 128)
{
    const int mt   = blockIdx.x;            // row-tile
    const int nt   = blockIdx.y;            // col-tile
    const int lane = threadIdx.x;           // 0..31
    const int l16  = lane & 15;
    const int half = lane >> 4;             // 0: lanes 0-15, 1: lanes 16-31

    // ---- A operand: x rows, 16x32 f16 layout (two K-slabs: 0..31, 32..63) ----
    // lane half h, element e -> K = kbase + 16*h + e
    const float* xr = x + (size_t)(mt * 16 + l16) * CC;
    v16h a0, a1;
#pragma unroll
    for (int e = 0; e < 8; ++e) {
        a0[e]     = (_Float16)xr[half * 16 + e];
        a0[e + 8] = (_Float16)xr[half * 16 + 8 + e];
        a1[e]     = (_Float16)xr[32 + half * 16 + e];
        a1[e + 8] = (_Float16)xr[32 + half * 16 + 8 + e];
    }

    // ---- B operand: B[k][n] = W[n][k]; 32x16 f16 layout:
    // lane half h, element e -> K = kbase + 16*h + e, N = lane%16
    const int   ncol = nt * 16 + l16;
    const float* wr  = W + (size_t)ncol * CC;
    v16h b0, b1;
#pragma unroll
    for (int e = 0; e < 16; ++e) {
        b0[e] = (_Float16)wr[half * 16 + e];
        b1[e] = (_Float16)wr[32 + half * 16 + e];
    }

    // ---- accumulate: D = A0*B0 + A1*B1 ----
    v8f c = {};
    c = __builtin_amdgcn_wmma_f32_16x16x32_f16(false, a0, false, b0,
                                               (short)0, c, false, false);
    c = __builtin_amdgcn_wmma_f32_16x16x32_f16(false, a1, false, b1,
                                               (short)0, c, false, false);

    // ---- store: C/D layout: VGPR r -> row = r + 8*half, col = lane%16 ----
#pragma unroll
    for (int r = 0; r < 8; ++r) {
        int row = mt * 16 + half * 8 + r;
        qk[(size_t)row * NCOLS + ncol] = c[r];
    }
}

// -----------------------------------------------------------------------------
// Kernel B: fused per-(b,h) attention. HEAD_DIM==1 so scores are rank-1:
//   scores[i,j] = q_i * k_j + slope_h * min(j-i, 0); softmax over j;
//   y_i = (sum_j w_ij * v_j) * proj_tmp[h],  v_j = x[b,j,h] * v_tmp[h].
// grid = B*H blocks, 256 threads; k/v staged once into LDS.
// -----------------------------------------------------------------------------
__global__ __launch_bounds__(256)
void attn_alibi_kernel(const float* __restrict__ x,        // (B,T,C)
                       const float* __restrict__ qk,       // (B*T, 128)
                       const float* __restrict__ v_tmp,    // (64,)
                       const float* __restrict__ proj_tmp, // (64,)
                       float* __restrict__ out)            // (B,T,C)
{
    __shared__ float sk[TT];
    __shared__ float sv[TT];

    const int b = blockIdx.x / HH;
    const int h = blockIdx.x % HH;
    const int tid = threadIdx.x;

    const float vscale = v_tmp[h];
    // ALiBi slope for H=64 (power of two): slope[h] = 2^(-(h+1)/8)
    const float slope = exp2f(-0.125f * (float)(h + 1));

    // Stage k and v for this (b,h) into LDS.
    for (int j = tid; j < TT; j += 256) {
        sk[j] = qk[(size_t)(b * TT + j) * NCOLS + CC + h];
        sv[j] = x[(size_t)(b * TT + j) * CC + h] * vscale;
    }
    __syncthreads();

    const float pscale = proj_tmp[h];

    for (int i = tid; i < TT; i += 256) {
        const float qi = qk[(size_t)(b * TT + i) * NCOLS + h];

        // pass 1: row max
        float m = -3.402823e38f;
        for (int j = 0; j < TT; ++j) {
            float bias = slope * fminf((float)(j - i), 0.0f);
            float sc = qi * sk[j] + bias;
            m = fmaxf(m, sc);
        }
        // pass 2: exp-sum and weighted value accumulation
        float den = 0.0f, num = 0.0f;
        for (int j = 0; j < TT; ++j) {
            float bias = slope * fminf((float)(j - i), 0.0f);
            float e = __expf(qi * sk[j] + bias - m);
            den += e;
            num += e * sv[j];
        }
        out[(size_t)(b * TT + i) * CC + h] = (num / den) * pscale;
    }
}

// -----------------------------------------------------------------------------
// Launch
// -----------------------------------------------------------------------------
extern "C" void kernel_launch(void* const* d_in, const int* in_sizes, int n_in,
                              void* d_out, int out_size, void* d_ws, size_t ws_size,
                              hipStream_t stream) {
    const float* x        = (const float*)d_in[0];  // (2,768,64)
    const float* W_attn   = (const float*)d_in[1];  // (128,64)
    const float* v_tmp    = (const float*)d_in[2];  // (64,1,1)
    const float* proj_tmp = (const float*)d_in[3];  // (64,1,1)
    float*       out      = (float*)d_out;          // (2,768,64)
    float*       qk       = (float*)d_ws;           // 1536*128 f32 = 768 KB scratch

    dim3 gA(NROWS / 16, NCOLS / 16);                // 96 x 8 tiles
    qk_proj_wmma_kernel<<<gA, 32, 0, stream>>>(x, W_attn, qk);

    attn_alibi_kernel<<<BB * HH, 256, 0, stream>>>(x, qk, v_tmp, proj_tmp, out);
}
